// PAGTN_80333068304728
// MI455X (gfx1250) — matrix operand order repaired and checked by reading, not compiled, not run on hardware
//
#include <hip/hip_runtime.h>
#include <hip/hip_bf16.h>

#define V_   16384
#define E_   131072
#define B_   512
#define H_   5
#define DH_  64
#define NI_  128
#define HD_  320      /* H_*DH_ */
#define DIN_ 192      /* NI_+DH_ */
#define RD_  1024

typedef __attribute__((ext_vector_type(16))) _Float16 v16h;
typedef __attribute__((ext_vector_type(8)))  float    v8f;

// ---- ordered-uint encoding so atomicMax(uint) implements float segment-max ----
__device__ __forceinline__ unsigned f2ord(float f) {
    unsigned u = __float_as_uint(f);
    return (u & 0x80000000u) ? ~u : (u | 0x80000000u);
}
__device__ __forceinline__ float ord2f(unsigned u) {
    return __uint_as_float((u & 0x80000000u) ? (u & 0x7fffffffu) : ~u);
}

// =====================================================================
// WMMA GEMM:  C[z] = act( A[z] @ W + bias )  (optionally += old C)
// A: (M,K) f32 row-major, lda, batch offset sA; W: (K,N) f32 row-major.
// Block = 128 threads = 4 waves. Block tile: 64 (M) x 32 (N).
// Wave tile: 16 x 32 -> two 16x16 f32 accumulators, A fragment reused.
// K loop steps by 64 -> 4 x v_wmma_f32_16x16x32_f16 per iteration.
// B chunk (64K x 32N) staged once per block in LDS, pre-swizzled into the
// exact per-lane fragment layout (16 contiguous halves per lane, 32B aligned).
// Requires: K % 64 == 0, N % 32 == 0, M % 64 == 0 (true for all call sites).
// act: 0 none, 1 relu, 2 prelu(alphaPtr[0])
// =====================================================================
__global__ void k_gemm(const float* __restrict__ A, int lda, int sA,
                       const float* __restrict__ W, int ldw,
                       const float* __restrict__ bias,
                       float* __restrict__ C, int ldc, int sC,
                       int K, int act, int accum,
                       const float* __restrict__ alphaPtr)
{
    __shared__ _Float16 ldsB[2][2][32][16];   // [kchunk][nsub][lane][half] = 4 KB

    const int tid  = threadIdx.x;
    const int lane = tid & 31;
    const int wave = tid >> 5;
    const int z = blockIdx.z;

    const float* Ab = A + (size_t)z * sA;
    float*       Cb = C + (size_t)z * sC;

    const int n0 = blockIdx.x * 32;                 // block N tile
    const int m0 = (blockIdx.y * 4 + wave) * 16;    // wave M tile

    const int hi   = lane >> 4;     // half-wave select
    const int l16  = lane & 15;
    const int arow = m0 + l16;

    // LDS fill mapping: this thread owns column (n0+bn), k rows kb..kb+15
    const int bn = tid & 31;
    const int kb = (tid >> 5) * 16;         // 0,16,32,48
    const int f_kc = kb >> 5;               // k chunk (0/1)
    const int f_hi = (kb >> 4) & 1;         // half-wave row inside chunk
    const int f_t  = bn >> 4;               // n subtile
    _Float16* ldst = &ldsB[f_kc][f_t][f_hi * 16 + (bn & 15)][0];

    v8f acc0 = {};
    v8f acc1 = {};

    for (int k0 = 0; k0 < K; k0 += 64) {
        // ---- cooperative B stage: (k0..k0+63) x (n0..n0+31), f32 -> f16 ----
        {
            const float* gsrc = W + (size_t)(k0 + kb) * ldw + (n0 + bn);
            #pragma unroll
            for (int j = 0; j < 16; ++j)
                ldst[j] = (_Float16)gsrc[(size_t)j * ldw];
        }
        __syncthreads();

        // ---- A fragments for both 32-wide k chunks (16B-aligned b128 loads) ----
        v16h a0, a1;
        #pragma unroll
        for (int kc = 0; kc < 2; ++kc) {
            const float4* ap =
                (const float4*)(Ab + (size_t)arow * lda + k0 + kc * 32 + hi * 16);
            if (k0 + 64 < K)
                __builtin_prefetch((const void*)(ap + 16), 0, 1);
            float4 f0 = ap[0], f1 = ap[1];   // K = base+0..7
            float4 f2 = ap[2], f3 = ap[3];   // K = base+8..15
            v16h a;
            a[0]  = (_Float16)f0.x; a[1]  = (_Float16)f0.y;
            a[2]  = (_Float16)f0.z; a[3]  = (_Float16)f0.w;
            a[4]  = (_Float16)f1.x; a[5]  = (_Float16)f1.y;
            a[6]  = (_Float16)f1.z; a[7]  = (_Float16)f1.w;
            a[8]  = (_Float16)f2.x; a[9]  = (_Float16)f2.y;
            a[10] = (_Float16)f2.z; a[11] = (_Float16)f2.w;
            a[12] = (_Float16)f3.x; a[13] = (_Float16)f3.y;
            a[14] = (_Float16)f3.z; a[15] = (_Float16)f3.w;
            if (kc == 0) a0 = a; else a1 = a;
        }

        // ---- B fragments from LDS (32B-aligned v16h loads) ----
        v16h b00 = *(const v16h*)&ldsB[0][0][lane][0];
        v16h b01 = *(const v16h*)&ldsB[0][1][lane][0];
        v16h b10 = *(const v16h*)&ldsB[1][0][lane][0];
        v16h b11 = *(const v16h*)&ldsB[1][1][lane][0];

        acc0 = __builtin_amdgcn_wmma_f32_16x16x32_f16(false, a0, false, b00,
                                                      (short)0, acc0, false, false);
        acc1 = __builtin_amdgcn_wmma_f32_16x16x32_f16(false, a0, false, b01,
                                                      (short)0, acc1, false, false);
        acc0 = __builtin_amdgcn_wmma_f32_16x16x32_f16(false, a1, false, b10,
                                                      (short)0, acc0, false, false);
        acc1 = __builtin_amdgcn_wmma_f32_16x16x32_f16(false, a1, false, b11,
                                                      (short)0, acc1, false, false);
        __syncthreads();
    }

    const float alpha = (act == 2) ? alphaPtr[0] : 0.f;
    #pragma unroll
    for (int t = 0; t < 2; ++t) {
        const v8f acc = t ? acc1 : acc0;
        const int bcol = n0 + t * 16 + l16;
        const float bv = bias[bcol];
        #pragma unroll
        for (int r = 0; r < 8; ++r) {
            int m = m0 + r + hi * 8;           // D layout: VGPR r -> rows r, r+8
            size_t ci = (size_t)m * ldc + bcol;
            float v = acc[r] + bv;
            if (accum) v += Cb[ci];
            if (act == 1)      v = fmaxf(v, 0.f);
            else if (act == 2) v = (v >= 0.f) ? v : alpha * v;
            Cb[ci] = v;
        }
    }
}

// ---------------- utility fill ----------------
__global__ void k_fill(unsigned* __restrict__ p, unsigned val, long n) {
    long i = (long)blockIdx.x * blockDim.x + threadIdx.x;
    if (i < n) p[i] = val;
}

// =====================================================================
// Edge attention score: one wave per (edge, head).
// score = dot(relu(asrc[s]+adst[d]+aedg[e]), Wdot) + bdot ; atomicMax seg-max.
// =====================================================================
__global__ void k_edge_score(const float* __restrict__ asrc,
                             const float* __restrict__ adst,
                             const float* __restrict__ aedg,
                             const float* __restrict__ wdot,
                             const float* __restrict__ bdot,
                             const int* __restrict__ src,
                             const int* __restrict__ dst,
                             float* __restrict__ score,
                             unsigned* __restrict__ smax)
{
    int pair = (int)((blockIdx.x * blockDim.x + threadIdx.x) >> 5);
    int lane = threadIdx.x & 31;
    if (pair >= E_ * H_) return;
    int e = pair / H_, hh = pair - e * H_;
    int s = src[e], d = dst[e];
    const float* ps = asrc + (size_t)s * HD_ + hh * DH_;
    const float* pd = adst + (size_t)d * HD_ + hh * DH_;
    const float* pe = aedg + (size_t)e * DH_;
    float acc = 0.f;
    #pragma unroll
    for (int t = 0; t < 2; ++t) {
        int j = lane + t * 32;
        float v = fmaxf(ps[j] + pd[j] + pe[j], 0.f);
        acc += v * wdot[j];
    }
    #pragma unroll
    for (int off = 16; off > 0; off >>= 1) acc += __shfl_xor(acc, off, 32);
    if (lane == 0) {
        acc += bdot[0];
        score[pair] = acc;
        atomicMax(&smax[(size_t)d * H_ + hh], f2ord(acc));
    }
}

__global__ void k_edge_expsum(float* __restrict__ score,
                              const unsigned* __restrict__ smax,
                              float* __restrict__ ssum,
                              const int* __restrict__ dst)
{
    int pair = blockIdx.x * blockDim.x + threadIdx.x;
    if (pair >= E_ * H_) return;
    int e = pair / H_, hh = pair - e * H_;
    int d = dst[e];
    float ex = expf(score[pair] - ord2f(smax[(size_t)d * H_ + hh]));
    score[pair] = ex;
    atomicAdd(&ssum[(size_t)d * H_ + hh], ex);
}

// msg = attn * (mdst[src] + medg); scatter-add into xnext[dst] (pre-loaded with wgt_n term)
__global__ void k_edge_message(const float* __restrict__ score,
                               const float* __restrict__ ssum,
                               const float* __restrict__ mdst,
                               const float* __restrict__ medg,
                               const int* __restrict__ src,
                               const int* __restrict__ dst,
                               float* __restrict__ xnext)
{
    int idx = blockIdx.x * blockDim.x + threadIdx.x;   // E_*H_*DH_ exactly
    int j = idx & 63;
    int pair = idx >> 6;
    int e = pair / H_, hh = pair - e * H_;
    int s = src[e], d = dst[e];
    float attn = score[pair] / ssum[(size_t)d * H_ + hh];
    float m = attn * (mdst[(size_t)s * HD_ + hh * DH_ + j] +
                      medg[(size_t)e * DH_ + j]);
    atomicAdd(&xnext[(size_t)d * HD_ + hh * DH_ + j], m);
}

// feat = concat(node_feats, mean_over_heads(x))
__global__ void k_mean_concat(const float* __restrict__ nf,
                              const float* __restrict__ x,
                              float* __restrict__ feat)
{
    int idx = blockIdx.x * blockDim.x + threadIdx.x;
    if (idx >= V_ * DIN_) return;
    int v = idx / DIN_, t = idx - v * DIN_;
    float val;
    if (t < NI_) {
        val = nf[(size_t)v * NI_ + t];
    } else {
        int j = t - NI_;
        float s = 0.f;
        #pragma unroll
        for (int hh = 0; hh < H_; ++hh) s += x[(size_t)v * HD_ + hh * DH_ + j];
        val = s * (1.0f / H_);
    }
    feat[idx] = val;
}

// LSTM cell update from gates (i,f,g,o order)
__global__ void k_lstm_update(const float* __restrict__ gates,
                              float* __restrict__ c, float* __restrict__ h)
{
    int idx = blockIdx.x * blockDim.x + threadIdx.x;
    if (idx >= B_ * DIN_) return;
    int b = idx / DIN_, j = idx - b * DIN_;
    const float* g = gates + (size_t)b * 4 * DIN_;
    float gi = g[j], gf = g[DIN_ + j], gg = g[2 * DIN_ + j], go = g[3 * DIN_ + j];
    float si = 1.f / (1.f + expf(-gi));
    float sf = 1.f / (1.f + expf(-gf));
    float so = 1.f / (1.f + expf(-go));
    float cv = sf * c[idx] + si * tanhf(gg);
    c[idx] = cv;
    h[idx] = so * tanhf(cv);
}

// e[v] = dot(feat[v], h[gid[v]]) ; wave per node, fused segment-max
__global__ void k_s2s_dot(const float* __restrict__ feat,
                          const float* __restrict__ h,
                          const int* __restrict__ gid,
                          float* __restrict__ ev, unsigned* __restrict__ gmax)
{
    int v = (int)((blockIdx.x * blockDim.x + threadIdx.x) >> 5);
    int lane = threadIdx.x & 31;
    if (v >= V_) return;
    int g = gid[v];
    const float* fv = feat + (size_t)v * DIN_;
    const float* hv = h + (size_t)g * DIN_;
    float acc = 0.f;
    for (int j = lane; j < DIN_; j += 32) acc += fv[j] * hv[j];
    #pragma unroll
    for (int off = 16; off > 0; off >>= 1) acc += __shfl_xor(acc, off, 32);
    if (lane == 0) { ev[v] = acc; atomicMax(&gmax[g], f2ord(acc)); }
}

__global__ void k_s2s_expsum(float* __restrict__ ev,
                             const unsigned* __restrict__ gmax,
                             float* __restrict__ gsum,
                             const int* __restrict__ gid)
{
    int v = blockIdx.x * blockDim.x + threadIdx.x;
    if (v >= V_) return;
    int g = gid[v];
    float ex = expf(ev[v] - ord2f(gmax[g]));
    ev[v] = ex;
    atomicAdd(&gsum[g], ex);
}

__global__ void k_s2s_wsum(const float* __restrict__ ev,
                           const float* __restrict__ gsum,
                           const float* __restrict__ feat,
                           const int* __restrict__ gid,
                           float* __restrict__ r)
{
    int idx = blockIdx.x * blockDim.x + threadIdx.x;
    if (idx >= V_ * DIN_) return;
    int v = idx / DIN_, j = idx - v * DIN_;
    int g = gid[v];
    float alpha = ev[v] / gsum[g];
    atomicAdd(&r[(size_t)g * DIN_ + j], alpha * feat[idx]);
}

__global__ void k_s2s_qstar(const float* __restrict__ h,
                            const float* __restrict__ r,
                            float* __restrict__ qstar)
{
    int idx = blockIdx.x * blockDim.x + threadIdx.x;
    if (idx >= B_ * DIN_) return;
    int b = idx / DIN_, j = idx - b * DIN_;
    qstar[(size_t)b * 2 * DIN_ + j]        = h[idx];
    qstar[(size_t)b * 2 * DIN_ + DIN_ + j] = r[idx];
}

// =====================================================================
extern "C" void kernel_launch(void* const* d_in, const int* in_sizes, int n_in,
                              void* d_out, int out_size, void* d_ws, size_t ws_size,
                              hipStream_t stream)
{
    (void)in_sizes; (void)n_in; (void)out_size; (void)ws_size;

    // ---- inputs in setup_inputs() dict (insertion) order ----
    const float* node_feats = (const float*)d_in[0];
    const float* edge_feats = (const float*)d_in[1];
    const int*   src        = (const int*)d_in[2];
    const int*   dst        = (const int*)d_in[3];
    const int*   gid        = (const int*)d_in[4];
    const float* aiW        = (const float*)d_in[5];
    const float* aiB        = (const float*)d_in[6];
    // layer L, slot j: attn_src{W,b}, attn_dst{W,b}, attn_edg{W,b}, attn_dot{W,b},
    //                  msg_src{W,b}, msg_dst{W,b}, msg_edg{W,b}, wgt_n{W,b}
    auto LP = [&](int L, int j) { return (const float*)d_in[7 + 16 * L + j]; };
    const float* Wih   = (const float*)d_in[55];
    const float* Whh   = (const float*)d_in[56];
    const float* bih   = (const float*)d_in[57];
    const float* bhh   = (const float*)d_in[58];
    const float* spW   = (const float*)d_in[59];
    const float* spB   = (const float*)d_in[60];
    const float* prelu = (const float*)d_in[61];

    // ---- workspace layout ----
    float* w = (float*)d_ws;
    size_t off = 0;
    auto alloc = [&](size_t n) { float* p = w + off; off += n; return p; };
    float* x0    = alloc((size_t)V_ * HD_);
    float* x1    = alloc((size_t)V_ * HD_);
    float* asrc  = alloc((size_t)V_ * HD_);
    float* adst  = alloc((size_t)V_ * HD_);
    float* mdst  = alloc((size_t)V_ * HD_);
    float* aedg  = alloc((size_t)E_ * DH_);
    float* medg  = alloc((size_t)E_ * DH_);
    float* score = alloc((size_t)E_ * H_);
    float* smaxF = alloc((size_t)V_ * H_);   // smax,ssum contiguous for one fill
    float* ssum  = alloc((size_t)V_ * H_);
    float* feat  = alloc((size_t)V_ * DIN_);
    float* gates = alloc((size_t)B_ * 4 * DIN_);
    float* hbuf  = alloc((size_t)B_ * DIN_); // hbuf,cbuf,qstar contiguous
    float* cbuf  = alloc((size_t)B_ * DIN_);
    float* qstar = alloc((size_t)B_ * 2 * DIN_);
    float* rbuf  = alloc((size_t)B_ * DIN_); // rbuf,gmax,gsum contiguous
    float* gmaxF = alloc((size_t)B_);
    float* gsum  = alloc((size_t)B_);
    float* ev    = alloc((size_t)V_);
    unsigned* smax = (unsigned*)smaxF;
    unsigned* gmax = (unsigned*)gmaxF;

    auto gemm = [&](const float* A, int lda, int sA, const float* W_, int ldw,
                    const float* bias, float* C, int ldc, int sC,
                    int M, int N, int K, int nb, int act, int accum,
                    const float* alpha) {
        dim3 g(N / 32, M / 64, nb);
        k_gemm<<<g, 128, 0, stream>>>(A, lda, sA, W_, ldw, bias, C, ldc, sC,
                                      K, act, accum, alpha);
    };
    auto fill = [&](void* p, unsigned val, size_t n) {
        k_fill<<<(unsigned)((n + 255) / 256), 256, 0, stream>>>((unsigned*)p, val, (long)n);
    };

    // ---- atom input projection + relu: (V,128)@(128,320) ----
    gemm(node_feats, NI_, 0, aiW, HD_, aiB, x0, HD_, 0, V_, HD_, NI_, 1, 1, 0, nullptr);

    float* cur = x0;
    float* nxt = x1;
    for (int L = 0; L < 3; ++L) {
        // head-batched (z = head): (V,64)@(64,64), shared W across heads
        gemm(cur, HD_, DH_, LP(L, 0),  DH_, LP(L, 1),  asrc, HD_, DH_, V_, DH_, DH_, H_, 0, 0, nullptr);
        gemm(cur, HD_, DH_, LP(L, 2),  DH_, LP(L, 3),  adst, HD_, DH_, V_, DH_, DH_, H_, 0, 0, nullptr);
        gemm(cur, HD_, DH_, LP(L, 10), DH_, LP(L, 11), mdst, HD_, DH_, V_, DH_, DH_, H_, 0, 0, nullptr);
        gemm(cur, HD_, DH_, LP(L, 14), DH_, LP(L, 15), nxt,  HD_, DH_, V_, DH_, DH_, H_, 0, 0, nullptr);
        // edge projections: (E,64)@(64,64)
        gemm(edge_feats, DH_, 0, LP(L, 4),  DH_, LP(L, 5),  aedg, DH_, 0, E_, DH_, DH_, 1, 0, 0, nullptr);
        gemm(edge_feats, DH_, 0, LP(L, 12), DH_, LP(L, 13), medg, DH_, 0, E_, DH_, DH_, 1, 0, 0, nullptr);
        // segment softmax over dst + message scatter into nxt
        fill(smax, 0u, (size_t)V_ * H_ * 2);   // -inf(ordered) max + zero sums
        k_edge_score<<<E_ * H_ / 8, 256, 0, stream>>>(asrc, adst, aedg, LP(L, 6), LP(L, 7),
                                                      src, dst, score, smax);
        k_edge_expsum<<<(E_ * H_ + 255) / 256, 256, 0, stream>>>(score, smax, ssum, dst);
        k_edge_message<<<(E_ * H_ * DH_) / 256, 256, 0, stream>>>(score, ssum, mdst, medg,
                                                                  src, dst, nxt);
        float* t = cur; cur = nxt; nxt = t;
    }

    k_mean_concat<<<(V_ * DIN_ + 255) / 256, 256, 0, stream>>>(node_feats, cur, feat);

    // ---- Set2Set (3 iterations) ----
    fill(hbuf, 0u, (size_t)B_ * DIN_ * 2 + (size_t)B_ * 2 * DIN_); // h, c, q_star = 0
    for (int it = 0; it < 3; ++it) {
        gemm(qstar, 2 * DIN_, 0, Wih, 4 * DIN_, bih, gates, 4 * DIN_, 0,
             B_, 4 * DIN_, 2 * DIN_, 1, 0, 0, nullptr);
        gemm(hbuf, DIN_, 0, Whh, 4 * DIN_, bhh, gates, 4 * DIN_, 0,
             B_, 4 * DIN_, DIN_, 1, 0, 1, nullptr);                 // accumulate
        k_lstm_update<<<(B_ * DIN_ + 255) / 256, 256, 0, stream>>>(gates, cbuf, hbuf);
        fill(rbuf, 0u, (size_t)B_ * DIN_ + 2 * B_);                 // r, gmax, gsum
        k_s2s_dot<<<V_ / 8, 256, 0, stream>>>(feat, hbuf, gid, ev, gmax);
        k_s2s_expsum<<<(V_ + 255) / 256, 256, 0, stream>>>(ev, gmax, gsum, gid);
        k_s2s_wsum<<<(V_ * DIN_ + 255) / 256, 256, 0, stream>>>(ev, gsum, feat, gid, rbuf);
        k_s2s_qstar<<<(B_ * DIN_ + 255) / 256, 256, 0, stream>>>(hbuf, rbuf, qstar);
    }

    // ---- sparsify + PReLU -> d_out (512,1024) ----
    gemm(qstar, 2 * DIN_, 0, spW, RD_, spB, (float*)d_out, RD_, 0,
         B_, RD_, 2 * DIN_, 1, 2, 0, prelu);
}
